// LossKMeansWasserstein_7627861918346
// MI455X (gfx1250) — matrix operand order repaired and checked by reading, not compiled, not run on hardware
//
#include <hip/hip_runtime.h>
#include <cmath>

typedef __attribute__((ext_vector_type(16))) _Float16 v16h;
typedef __attribute__((ext_vector_type(8)))  float    v8f;

namespace {
constexpr int    NP = 3072, MP = 3072, DD = 64, KC = 8, NB = 12, NIT = 30;
constexpr float  EPS = 0.0025f, SC2 = 0.64f;
constexpr size_t NM = (size_t)NP * MP;
// ---- workspace layout (units: 4-byte words) ----
constexpr size_t O_CXY = 0, O_CXX = NM, O_CYY = 2 * NM;
constexpr size_t O_XN   = 3 * NM;
constexpr size_t O_YN   = O_XN + NP;
constexpr size_t O_LA   = O_YN + MP;                 // loga [K][N]
constexpr size_t O_LB   = O_LA + (size_t)KC * NP;    // logb [K][M]
constexpr size_t O_PRED = O_LB + (size_t)KC * MP;    // pred_x [N] (int)
constexpr size_t O_Z0   = O_PRED + NP;               // ---- zeroed region ----
constexpr size_t O_CNTX = O_Z0;
constexpr size_t O_CNTY = O_CNTX + KC;
constexpr size_t O_MAXC = O_CNTY + KC;               // 3 ints (float bits)
constexpr size_t O_FILP = O_MAXC + 3;                // [NB][K]
constexpr size_t O_DOTP = O_FILP + (size_t)NB * KC;  // [6][K][NB]
constexpr size_t O_FXY  = O_DOTP + (size_t)6 * KC * NB;
constexpr size_t O_GXY  = O_FXY + (size_t)KC * NP;
constexpr size_t O_FXX  = O_GXY + (size_t)KC * MP;
constexpr size_t O_GXX  = O_FXX + (size_t)KC * NP;
constexpr size_t O_FYY  = O_GXX + (size_t)KC * NP;
constexpr size_t O_GYY  = O_FYY + (size_t)KC * MP;
constexpr size_t O_ZEND = O_GYY + (size_t)KC * MP;   // ---- end zeroed ----
constexpr size_t O_F2XY = O_ZEND;
constexpr size_t O_G2XY = O_F2XY + (size_t)KC * NP;
constexpr size_t O_F2XX = O_G2XY + (size_t)KC * MP;
constexpr size_t O_G2XX = O_F2XX + (size_t)KC * NP;
constexpr size_t O_F2YY = O_G2XX + (size_t)KC * NP;
constexpr size_t O_G2YY = O_F2YY + (size_t)KC * MP;
} // namespace

__global__ void k_zero(unsigned* p, unsigned long long n) {
  unsigned long long i = (unsigned long long)blockIdx.x * 256ull + threadIdx.x;
  if (i < n) p[i] = 0u;
}

__global__ __launch_bounds__(256) void k_norms(const float* __restrict__ X,
                                               const float* __restrict__ Y,
                                               float* __restrict__ xn,
                                               float* __restrict__ yn) {
  int i = blockIdx.x * 256 + threadIdx.x;
  if (i < NP) {
    const float* r = X + (size_t)i * DD;
    float s = 0.f;
    for (int d = 0; d < DD; ++d) s += r[d] * r[d];
    xn[i] = 0.5f * s;
  } else if (i < NP + MP) {
    int j = i - NP;
    const float* r = Y + (size_t)j * DD;
    float s = 0.f;
    for (int d = 0; d < DD; ++d) s += r[d] * r[d];
    yn[j] = 0.5f * s;
  }
}

// d_x to 8 centers, argmin -> pred, softmax(-d) partial sums -> fillp[block][k]
__global__ __launch_bounds__(256) void k_assign(const float* __restrict__ x,
                                                const float* __restrict__ cen,
                                                float* __restrict__ fillp,
                                                int* __restrict__ pred) {
  __shared__ float cs[KC * DD];
  for (int t = threadIdx.x; t < KC * DD; t += 256) cs[t] = cen[t];
  __syncthreads();
  int i = blockIdx.x * 256 + threadIdx.x;
  float d[KC];
#pragma unroll
  for (int k = 0; k < KC; ++k) d[k] = 0.f;
  const float* xr = x + (size_t)i * DD;
  for (int dd = 0; dd < DD; ++dd) {
    float xv = xr[dd];
#pragma unroll
    for (int k = 0; k < KC; ++k) { float df = xv - cs[k * DD + dd]; d[k] += df * df; }
  }
  int am = 0; float dm = d[0];
#pragma unroll
  for (int k = 1; k < KC; ++k) if (d[k] < dm) { dm = d[k]; am = k; }
  pred[i] = am;
  float e[KC], se = 0.f;
#pragma unroll
  for (int k = 0; k < KC; ++k) { e[k] = __expf(dm - d[k]); se += e[k]; }  // softmax(-d)
  float inv = 1.f / se;
  __shared__ float red[256];
  for (int k = 0; k < KC; ++k) {
    red[threadIdx.x] = e[k] * inv;
    __syncthreads();
    for (int o = 128; o > 0; o >>= 1) {
      if (threadIdx.x < o) red[threadIdx.x] += red[threadIdx.x + o];
      __syncthreads();
    }
    if (threadIdx.x == 0) fillp[(size_t)blockIdx.x * KC + k] = red[0];
    __syncthreads();
  }
}

__global__ void k_hist(const int* __restrict__ pred, int n, int* __restrict__ cnt) {
  int i = blockIdx.x * 256 + threadIdx.x;
  if (i < n) atomicAdd(&cnt[pred[i]], 1);
}

__global__ __launch_bounds__(256) void k_weights(const int* __restrict__ pred,
                                                 const int* __restrict__ cnt, int n,
                                                 float* __restrict__ logw) {
  int k = blockIdx.y;
  int i = blockIdx.x * 256 + threadIdx.x;
  if (i >= n) return;
  int c = cnt[k];
  float lw;
  if (c > 0) lw = (pred[i] == k) ? -__logf((float)c) : -1e30f;  // -1e30 == "-inf" sentinel
  else       lw = -__logf((float)n);                            // empty-cluster fallback
  logw[(size_t)k * n + i] = lw;
}

// C[i,j] = xn[i] + yn[j] - x_i . y_j   via f16 hi/lo split WMMA (3 products/K-chunk)
__global__ __launch_bounds__(128) void k_cost_wmma(const float* __restrict__ X,
                                                   const float* __restrict__ Y,
                                                   const float* __restrict__ xn,
                                                   const float* __restrict__ yn,
                                                   float* __restrict__ C, int cols) {
  int lane = threadIdx.x;          // 0..31, full wave, EXEC all-ones
  int m = lane & 15, kh = lane >> 4;
  int ti = blockIdx.y * 4 + threadIdx.y;
  int i0 = ti * 16, j0 = blockIdx.x * 16;
  const float* xrow = X + (size_t)(i0 + m) * DD;
  const float* yrow = Y + (size_t)(j0 + m) * DD;
  v8f acc = {};
#pragma unroll
  for (int kc = 0; kc < 2; ++kc) {   // K = 64 in two 32-deep chunks
    int kb = kc * 32;
    v16h ah, al, bh, bl;
    // A (16x32): lane m row; elems 0..7 -> K = kb+kh*8+e, elems 8..15 -> K = kb+16+kh*8+e
#pragma unroll
    for (int e = 0; e < 8; ++e) {
      float v0 = xrow[kb + kh * 8 + e];
      float v1 = xrow[kb + 16 + kh * 8 + e];
      _Float16 h0 = (_Float16)v0, h1 = (_Float16)v1;
      ah[e] = h0;     al[e] = (_Float16)(v0 - (float)h0);
      ah[e + 8] = h1; al[e + 8] = (_Float16)(v1 - (float)h1);
    }
    // B (32x16): lane n=m col; elems 0..7 -> K = kb+kh*8+e, elems 8..15 -> K = kb+16+kh*8+(e-8)
#pragma unroll
    for (int e = 0; e < 16; ++e) {
      float v = yrow[kb + (e & 8) * 2 + kh * 8 + (e & 7)];
      _Float16 h = (_Float16)v;
      bh[e] = h; bl[e] = (_Float16)(v - (float)h);
    }
    acc = __builtin_amdgcn_wmma_f32_16x16x32_f16(false, ah, false, bh, (short)0, acc, false, false);
    acc = __builtin_amdgcn_wmma_f32_16x16x32_f16(false, ah, false, bl, (short)0, acc, false, false);
    acc = __builtin_amdgcn_wmma_f32_16x16x32_f16(false, al, false, bh, (short)0, acc, false, false);
  }
  int cj = j0 + m;
  float ynj = yn[cj];
#pragma unroll
  for (int r = 0; r < 8; ++r) {      // C/D layout: M = r + 8*kh, N = lane&15
    int ci = i0 + kh * 8 + r;
    C[(size_t)ci * cols + cj] = xn[ci] + ynj - acc[r];
  }
}

__global__ __launch_bounds__(256) void k_max(const float* __restrict__ C,
                                             unsigned long long n, int* __restrict__ out) {
  unsigned long long stride = (unsigned long long)gridDim.x * 256ull;
  float m = 0.f;  // costs are >= ~0; eps0 is EPS-clamped later anyway
  for (unsigned long long i = (unsigned long long)blockIdx.x * 256ull + threadIdx.x; i < n; i += stride)
    m = fmaxf(m, C[i]);
  __shared__ float red[256];
  red[threadIdx.x] = m; __syncthreads();
  for (int o = 128; o > 0; o >>= 1) {
    if (threadIdx.x < o) red[threadIdx.x] = fmaxf(red[threadIdx.x], red[threadIdx.x + o]);
    __syncthreads();
  }
  if (threadIdx.x == 0) atomicMax(out, __float_as_int(red[0]));  // exact for non-neg floats
}

// out[k][row] = -eps * lse_j( logw[k][j] + (vec[k][j] - C[row,j]) / eps ), masked entries skipped
__global__ __launch_bounds__(256) void k_row_lse(const float* __restrict__ C, int ncols,
                                                 const float* __restrict__ vec,
                                                 const float* __restrict__ logw,
                                                 const float* __restrict__ maxC, float pt,
                                                 float* __restrict__ outv, int nrows) {
  int row = blockIdx.x, k = blockIdx.y;
  float eps = fmaxf(fmaxf(maxC[0], EPS) * pt, EPS);
  float ie = 1.0f / eps;
  const float* cr = C + (size_t)row * ncols;
  const float* vk = vec + (size_t)k * ncols;
  const float* lk = logw + (size_t)k * ncols;
  float m = -INFINITY, s = 0.f;
  for (int j = threadIdx.x; j < ncols; j += 256) {
    float lb = lk[j];
    if (lb < -1e29f) continue;          // masked -> skip before the C load
    float val = lb + (vk[j] - cr[j]) * ie;
    if (val <= m) s += __expf(val - m);
    else { s = s * __expf(m - val) + 1.f; m = val; }
  }
  __shared__ float sm[256], ss[256];
  sm[threadIdx.x] = m; ss[threadIdx.x] = s; __syncthreads();
  for (int o = 128; o > 0; o >>= 1) {
    if (threadIdx.x < o) {
      float m1 = sm[threadIdx.x], m2 = sm[threadIdx.x + o];
      float M = fmaxf(m1, m2), S;
      if (M == -INFINITY) S = 0.f;
      else S = ss[threadIdx.x] * __expf(m1 - M) + ss[threadIdx.x + o] * __expf(m2 - M);
      sm[threadIdx.x] = M; ss[threadIdx.x] = S;
    }
    __syncthreads();
  }
  if (threadIdx.x == 0) outv[(size_t)k * nrows + row] = -eps * (sm[0] + __logf(ss[0]));
}

// column sweep for the non-symmetric Cxy: 32 cols/block, coalesced 128B row reads
__global__ __launch_bounds__(256) void k_col_lse(const float* __restrict__ C, int ncols, int nrows,
                                                 const float* __restrict__ vec,
                                                 const float* __restrict__ logw,
                                                 const float* __restrict__ maxC, float pt,
                                                 float* __restrict__ outv) {
  int tx = threadIdx.x & 31, ty = threadIdx.x >> 5;
  int j = blockIdx.x * 32 + tx, k = blockIdx.y;
  float eps = fmaxf(fmaxf(maxC[0], EPS) * pt, EPS);
  float ie = 1.0f / eps;
  const float* vk = vec + (size_t)k * nrows;
  const float* lk = logw + (size_t)k * nrows;
  float m = -INFINITY, s = 0.f;
  for (int i = ty; i < nrows; i += 8) {
    float lb = lk[i];
    if (lb < -1e29f) continue;
    float val = lb + (vk[i] - C[(size_t)i * ncols + j]) * ie;
    if (val <= m) s += __expf(val - m);
    else { s = s * __expf(m - val) + 1.f; m = val; }
  }
  __shared__ float sm[256], ss[256];
  int idx = ty * 32 + tx;
  sm[idx] = m; ss[idx] = s; __syncthreads();
  for (int o = 4; o > 0; o >>= 1) {
    if (ty < o) {
      float m1 = sm[idx], m2 = sm[idx + o * 32];
      float M = fmaxf(m1, m2), S;
      if (M == -INFINITY) S = 0.f;
      else S = ss[idx] * __expf(m1 - M) + ss[idx + o * 32] * __expf(m2 - M);
      sm[idx] = M; ss[idx] = S;
    }
    __syncthreads();
  }
  if (ty == 0) outv[(size_t)k * ncols + j] = -eps * (sm[tx] + __logf(ss[tx]));
}

// parts[k][block] = sum_i a_i * v[k][i]   (deterministic per-block tree)
__global__ __launch_bounds__(256) void k_dot(const float* __restrict__ v,
                                             const int* __restrict__ pred,
                                             const int* __restrict__ cnt, int n,
                                             float* __restrict__ parts) {
  int k = blockIdx.y;
  int i = blockIdx.x * 256 + threadIdx.x;
  int c = cnt[k];
  float a = (c > 0) ? ((pred[i] == k) ? 1.0f / (float)c : 0.0f) : 1.0f / (float)n;
  __shared__ float red[256];
  red[threadIdx.x] = a * v[(size_t)k * n + i];
  __syncthreads();
  for (int o = 128; o > 0; o >>= 1) {
    if (threadIdx.x < o) red[threadIdx.x] += red[threadIdx.x + o];
    __syncthreads();
  }
  if (threadIdx.x == 0) parts[(size_t)k * NB + blockIdx.x] = red[0];
}

__global__ void k_final(const float* __restrict__ fillp, const float* __restrict__ ftgt,
                        const float* __restrict__ dotp, const int* __restrict__ cntx,
                        const int* __restrict__ cnty, float* __restrict__ out) {
  if (threadIdx.x || blockIdx.x) return;
  float loss = 0.f;
  for (int k = 0; k < KC; ++k) {           // loss_fil = mean((filling_x - filling_target)^2)
    float fx = 0.f;
    for (int b = 0; b < NB; ++b) fx += fillp[(size_t)b * KC + k];
    fx *= (1.0f / (float)NP);
    float df = fx - ftgt[k];
    loss += df * df * (1.0f / (float)KC);
  }
  for (int k = 0; k < KC; ++k) {           // debiased sinkhorn divergence per cluster
    float s[6];
    for (int buf = 0; buf < 6; ++buf) {
      float t = 0.f;
      for (int b = 0; b < NB; ++b) t += dotp[(size_t)buf * KC * NB + (size_t)k * NB + b];
      s[buf] = t;
    }
    if (cntx[k] > 0 && cnty[k] > 0)
      loss += (s[0] + s[1]) - 0.5f * (s[2] + s[3]) - 0.5f * (s[4] + s[5]);
  }
  out[0] = loss;
}

extern "C" void kernel_launch(void* const* d_in, const int* in_sizes, int n_in,
                              void* d_out, int out_size, void* d_ws, size_t ws_size,
                              hipStream_t stream) {
  (void)in_sizes; (void)n_in; (void)out_size; (void)ws_size;
  const float* x     = (const float*)d_in[0];
  const float* tgt   = (const float*)d_in[1];
  const float* cen   = (const float*)d_in[2];
  const float* ftgt  = (const float*)d_in[3];
  const int*   predt = (const int*)d_in[4];
  float* w  = (float*)d_ws;
  int*   wi = (int*)d_ws;
  float* out = (float*)d_out;

  float* Cxy = w + O_CXY; float* Cxx = w + O_CXX; float* Cyy = w + O_CYY;
  float* xn = w + O_XN;  float* yn = w + O_YN;
  float* loga = w + O_LA; float* logb = w + O_LB;
  int* pred = wi + O_PRED;
  int* cntx = wi + O_CNTX; int* cnty = wi + O_CNTY;
  int* maxC = wi + O_MAXC; float* mc = (float*)maxC;
  float* fillp = w + O_FILP; float* dotp = w + O_DOTP;
  float* fxy = w + O_FXY; float* gxy = w + O_GXY;
  float* fxx = w + O_FXX; float* gxx = w + O_GXX;
  float* fyy = w + O_FYY; float* gyy = w + O_GYY;
  float* f2xy = w + O_F2XY; float* g2xy = w + O_G2XY;
  float* f2xx = w + O_F2XX; float* g2xx = w + O_G2XX;
  float* f2yy = w + O_F2YY; float* g2yy = w + O_G2YY;

  { // zero counters, parts, maxC and all potentials (f0 = g0 = 0)
    unsigned long long zn = (unsigned long long)(O_ZEND - O_Z0);
    k_zero<<<dim3((unsigned)((zn + 255) / 256)), dim3(256), 0, stream>>>((unsigned*)(wi + O_Z0), zn);
  }
  k_norms<<<dim3((NP + MP) / 256), dim3(256), 0, stream>>>(x, tgt, xn, yn);
  k_assign<<<dim3(NB), dim3(256), 0, stream>>>(x, cen, fillp, pred);
  k_hist<<<dim3(NB), dim3(256), 0, stream>>>(pred, NP, cntx);
  k_hist<<<dim3(NB), dim3(256), 0, stream>>>(predt, MP, cnty);
  k_weights<<<dim3(NB, KC), dim3(256), 0, stream>>>(pred, cntx, NP, loga);
  k_weights<<<dim3(NB, KC), dim3(256), 0, stream>>>(predt, cnty, MP, logb);

  dim3 gg(MP / 16, NP / 64), gb(32, 4);   // one 16x16 WMMA tile per wave, 4 waves/WG
  k_cost_wmma<<<gg, gb, 0, stream>>>(x, tgt, xn, yn, Cxy, MP);
  k_cost_wmma<<<gg, gb, 0, stream>>>(x, x, xn, xn, Cxx, NP);
  k_cost_wmma<<<gg, gb, 0, stream>>>(tgt, tgt, yn, yn, Cyy, MP);
  k_max<<<dim3(1024), dim3(256), 0, stream>>>(Cxy, NM, maxC + 0);
  k_max<<<dim3(1024), dim3(256), 0, stream>>>(Cxx, NM, maxC + 1);
  k_max<<<dim3(1024), dim3(256), 0, stream>>>(Cyy, NM, maxC + 2);

  float pt = 1.0f;                         // SCAL2^t
  for (int t = 0; t < NIT; ++t) {
    // Cxy: f = row-lse(g, logb); g = col-lse(f, loga)
    k_row_lse<<<dim3(NP, KC), dim3(256), 0, stream>>>(Cxy, MP, gxy, logb, mc + 0, pt, fxy, NP);
    k_col_lse<<<dim3(MP / 32, KC), dim3(256), 0, stream>>>(Cxy, MP, NP, fxy, loga, mc + 0, pt, gxy);
    // Cxx symmetric: both updates via row kernel
    k_row_lse<<<dim3(NP, KC), dim3(256), 0, stream>>>(Cxx, NP, gxx, loga, mc + 1, pt, fxx, NP);
    k_row_lse<<<dim3(NP, KC), dim3(256), 0, stream>>>(Cxx, NP, fxx, loga, mc + 1, pt, gxx, NP);
    // Cyy symmetric
    k_row_lse<<<dim3(MP, KC), dim3(256), 0, stream>>>(Cyy, MP, gyy, logb, mc + 2, pt, fyy, MP);
    k_row_lse<<<dim3(MP, KC), dim3(256), 0, stream>>>(Cyy, MP, fyy, logb, mc + 2, pt, gyy, MP);
    pt *= SC2;
  }
  // final differentiable update at eps = EPS (pt = 0 -> eps clamps to EPS)
  k_row_lse<<<dim3(NP, KC), dim3(256), 0, stream>>>(Cxy, MP, gxy, logb, mc + 0, 0.0f, f2xy, NP);
  k_col_lse<<<dim3(MP / 32, KC), dim3(256), 0, stream>>>(Cxy, MP, NP, fxy, loga, mc + 0, 0.0f, g2xy);
  k_row_lse<<<dim3(NP, KC), dim3(256), 0, stream>>>(Cxx, NP, gxx, loga, mc + 1, 0.0f, f2xx, NP);
  k_row_lse<<<dim3(NP, KC), dim3(256), 0, stream>>>(Cxx, NP, fxx, loga, mc + 1, 0.0f, g2xx, NP);
  k_row_lse<<<dim3(MP, KC), dim3(256), 0, stream>>>(Cyy, MP, gyy, logb, mc + 2, 0.0f, f2yy, MP);
  k_row_lse<<<dim3(MP, KC), dim3(256), 0, stream>>>(Cyy, MP, fyy, logb, mc + 2, 0.0f, g2yy, MP);
  // s = <a,f2> + <b,g2> per matrix/cluster
  k_dot<<<dim3(NB, KC), dim3(256), 0, stream>>>(f2xy, pred,  cntx, NP, dotp + 0 * KC * NB);
  k_dot<<<dim3(NB, KC), dim3(256), 0, stream>>>(g2xy, predt, cnty, MP, dotp + 1 * KC * NB);
  k_dot<<<dim3(NB, KC), dim3(256), 0, stream>>>(f2xx, pred,  cntx, NP, dotp + 2 * KC * NB);
  k_dot<<<dim3(NB, KC), dim3(256), 0, stream>>>(g2xx, pred,  cntx, NP, dotp + 3 * KC * NB);
  k_dot<<<dim3(NB, KC), dim3(256), 0, stream>>>(f2yy, predt, cnty, MP, dotp + 4 * KC * NB);
  k_dot<<<dim3(NB, KC), dim3(256), 0, stream>>>(g2yy, predt, cnty, MP, dotp + 5 * KC * NB);

  k_final<<<dim3(1), dim3(1), 0, stream>>>(fillp, ftgt, dotp, cntx, cnty, out);
}
